// ONLSTMStack_67345087201508
// MI455X (gfx1250) — matrix operand order repaired and checked by reading, not compiled, not run on hardware
//
#include <hip/hip_runtime.h>
#include <hip/hip_bf16.h>
#include <math.h>

typedef __attribute__((ext_vector_type(16))) _Float16 v16h;
typedef __attribute__((ext_vector_type(8)))  float    v8f;
typedef __attribute__((ext_vector_type(4)))  _Float16 v4h;
typedef __attribute__((ext_vector_type(4)))  int      v4i;

#define AS1 __attribute__((address_space(1)))
#define AS3 __attribute__((address_space(3)))

#define T_STEPS 256
#define BATCH   64
#define EMBD    1024
#define HIDN    1024
#define GG      4128      // 4*HID + 2*NCH
#define NCHK    16
#define NBLK    64        // persistent-kernel grid for recurrence

#if __has_builtin(__builtin_amdgcn_global_load_async_to_lds_b128) && \
    __has_builtin(__builtin_amdgcn_s_wait_asynccnt)
#define USE_ASYNC_LDS 1
#else
#define USE_ASYNC_LDS 0
#endif

__device__ __forceinline__ float sigm(float x) { return 1.0f / (1.0f + __expf(-x)); }

// ---------------------------------------------------------------------------
// Weight prep: W (K x N, f32 row-major) -> Wt (N x K, f16 row-major)
// ---------------------------------------------------------------------------
__global__ __launch_bounds__(256) void transpose_cast(const float* __restrict__ W,
                                                      _Float16* __restrict__ Wt,
                                                      int K, int N) {
  __shared__ float tile[32][33];
  const int n0 = blockIdx.x * 32;
  const int k0 = blockIdx.y * 32;
  for (int i = threadIdx.x; i < 1024; i += 256) {
    int k = i >> 5, n = i & 31;
    tile[k][n] = W[(size_t)(k0 + k) * N + (n0 + n)];
  }
  __syncthreads();
  for (int i = threadIdx.x; i < 1024; i += 256) {
    int n = i >> 5, k = i & 31;
    Wt[(size_t)(n0 + n) * K + (k0 + k)] = (_Float16)tile[k][n];
  }
}

// ---------------------------------------------------------------------------
// Embedding gather -> f16 activations, plus mask (tokens != 0) to d_out tail
// ---------------------------------------------------------------------------
__global__ __launch_bounds__(256) void embed_cast(const int* __restrict__ tokens,
                                                  const float* __restrict__ emb,
                                                  _Float16* __restrict__ x,
                                                  float* __restrict__ mask_out) {
  const int i   = blockIdx.x;               // t*B + b, 0..16383
  const int tok = tokens[i];
  const float4* src = (const float4*)(emb + (size_t)tok * EMBD);
  v4h*          dst = (v4h*)(x + (size_t)i * EMBD);
  float4 v = src[threadIdx.x];              // 256 threads * 4 = 1024 elems
  v4h h = { (_Float16)v.x, (_Float16)v.y, (_Float16)v.z, (_Float16)v.w };
  dst[threadIdx.x] = h;
  if (threadIdx.x == 0) mask_out[i] = (tok != 0) ? 1.0f : 0.0f;
}

// ---------------------------------------------------------------------------
// Input projection GEMM: C(M,G) = A(M,1024) * Bt(G,1024)^T + bias, f32 out.
// WG tile 128x48, 8 waves, each wave 16x48 (3 accumulators).
// Double-buffered LDS staging: async global->LDS DMA (ASYNCcnt) into buf^1
// overlaps WMMA on buf; one barrier per K-step.
// Fragment layouts follow CDNA5 ISA 7.12.2 (16-bit A 16x32 / B 32x16).
// ---------------------------------------------------------------------------
__global__ __launch_bounds__(256) void gemm_wmma(const _Float16* __restrict__ A,
                                                 const _Float16* __restrict__ Bt,
                                                 const float* __restrict__ bias,
                                                 float* __restrict__ C) {
  __shared__ __align__(16) _Float16 As[2][128][40];  // 80B rows, 16B aligned
  __shared__ __align__(16) _Float16 Bs[2][48][40];

  const int tid    = threadIdx.x;
  const int wave   = tid >> 5;
  const int lane   = tid & 31;
  const int lrow   = lane & 15;
  const int m_base = blockIdx.x * 128;
  const int n_base = blockIdx.y * 48;
  const int aLo    = (lane < 16) ? 0 : 8;    // A lanes 0-15: K 0..7 & 16..23
  const int bLo    = (lane < 16) ? 0 : 8;    // B lanes 0-15: K 0..7 & 16..23

  // fixed per-thread staging assignment
  const int sr = tid >> 2;             // 0..63
  const int ss = (tid & 3) * 8;        // segment offset in halves (16B)

  const _Float16* gA0 = A  + (size_t)(m_base + sr) * 1024 + ss;
  const _Float16* gA1 = A  + (size_t)(m_base + 64 + sr) * 1024 + ss;
  const _Float16* gB  = Bt + (size_t)(n_base + sr) * 1024 + ss;

  // far prefetch of the weight panel (speculative, GL2)
  __builtin_prefetch(gB, 0, 1);

  v8f acc0 = {}, acc1 = {}, acc2 = {};

#if USE_ASYNC_LDS
#define STAGE(buf, kk)                                                        \
  do {                                                                        \
    __builtin_amdgcn_global_load_async_to_lds_b128(                           \
        (AS1 v4i*)(gA0 + (kk)), (AS3 v4i*)&As[buf][sr][ss], 0, 0);            \
    __builtin_amdgcn_global_load_async_to_lds_b128(                           \
        (AS1 v4i*)(gA1 + (kk)), (AS3 v4i*)&As[buf][64 + sr][ss], 0, 0);       \
    if (tid < 192)                                                            \
      __builtin_amdgcn_global_load_async_to_lds_b128(                         \
          (AS1 v4i*)(gB + (kk)), (AS3 v4i*)&Bs[buf][sr][ss], 0, 0);           \
  } while (0)
#define STAGE_WAIT() __builtin_amdgcn_s_wait_asynccnt(0)
#else
#define STAGE(buf, kk)                                                        \
  do {                                                                        \
    float4 va0 = *(const float4*)(gA0 + (kk));                                \
    float4 va1 = *(const float4*)(gA1 + (kk));                                \
    float4 vb = {};                                                           \
    if (tid < 192) vb = *(const float4*)(gB + (kk));                          \
    *(float4*)&As[buf][sr][ss]      = va0;                                    \
    *(float4*)&As[buf][64 + sr][ss] = va1;                                    \
    if (tid < 192) *(float4*)&Bs[buf][sr][ss] = vb;                           \
  } while (0)
#define STAGE_WAIT() ((void)0)
#endif

  // prologue: fill buffer 0
  STAGE(0, 0);
  STAGE_WAIT();
  __syncthreads();

  for (int k0 = 0; k0 < 1024; k0 += 32) {
    const int cur = (k0 >> 5) & 1;
    if (k0 + 32 < 1024) STAGE(cur ^ 1, k0 + 32);  // DMA next tile into idle buf

    union { v16h v; float4 f[2]; } a, b;
    a.f[0] = *(const float4*)&As[cur][wave * 16 + lrow][aLo];
    a.f[1] = *(const float4*)&As[cur][wave * 16 + lrow][aLo + 16];

    b.f[0] = *(const float4*)&Bs[cur][lrow][bLo];
    b.f[1] = *(const float4*)&Bs[cur][lrow][bLo + 16];
    acc0 = __builtin_amdgcn_wmma_f32_16x16x32_f16(false, a.v, false, b.v,
                                                  (short)0, acc0, false, false);
    b.f[0] = *(const float4*)&Bs[cur][16 + lrow][bLo];
    b.f[1] = *(const float4*)&Bs[cur][16 + lrow][bLo + 16];
    acc1 = __builtin_amdgcn_wmma_f32_16x16x32_f16(false, a.v, false, b.v,
                                                  (short)0, acc1, false, false);
    b.f[0] = *(const float4*)&Bs[cur][32 + lrow][bLo];
    b.f[1] = *(const float4*)&Bs[cur][32 + lrow][bLo + 16];
    acc2 = __builtin_amdgcn_wmma_f32_16x16x32_f16(false, a.v, false, b.v,
                                                  (short)0, acc2, false, false);

    STAGE_WAIT();        // this wave's DMAs into buf^1 done
    __syncthreads();     // all waves done reading cur + all DMAs visible
  }
#undef STAGE
#undef STAGE_WAIT

  // C/D layout: VGPR r -> lanes 0-15 row M=r, lanes 16-31 row M=8+r, col=lane%16
  const int col    = n_base + lrow;
  const int rowOff = m_base + wave * 16 + ((lane < 16) ? 0 : 8);
  const float b0 = bias[col], b1 = bias[col + 16], b2 = bias[col + 32];
#pragma unroll
  for (int r = 0; r < 8; ++r) {
    size_t row = (size_t)(rowOff + r);
    C[row * GG + col]      = acc0[r] + b0;
    C[row * GG + col + 16] = acc1[r] + b1;
    C[row * GG + col + 32] = acc2[r] + b2;
  }
}

// ---------------------------------------------------------------------------
// Device-wide barrier for the persistent recurrence kernel (monotonic counter)
// ---------------------------------------------------------------------------
__device__ __forceinline__ void grid_sync(unsigned* bar, unsigned* phase) {
  __threadfence();
  __syncthreads();
  if (threadIdx.x == 0) {
    unsigned target = *phase + NBLK;
    atomicAdd(bar, 1u);
    while (__hip_atomic_load(bar, __ATOMIC_ACQUIRE, __HIP_MEMORY_SCOPE_AGENT) < target)
      __builtin_amdgcn_s_sleep(1);
  }
  *phase += NBLK;
  __syncthreads();
  __threadfence();
}

__global__ void zero_bar(unsigned* bar) { if (threadIdx.x == 0) *bar = 0u; }

// ---------------------------------------------------------------------------
// Persistent ON-LSTM recurrence: 256 steps of (64x1024 @ 1024x4128) + gates.
// Phase 1: WMMA GEMM into gates buffer. Each wave computes a 16x32 strip
// (two N-tiles share one A fragment): 516 strip-units over 512 waves.
// Phase 2: cumsoftmax + ON-LSTM pointwise (one block per batch row).
// ---------------------------------------------------------------------------
__global__ __launch_bounds__(256) void onlstm_recur(
    const float* __restrict__ t_in,      // (T*B, G) incl. b_ih
    const _Float16* __restrict__ Wt_hh,  // (G, HID) = W_hh^T f16
    const float* __restrict__ b_hh,      // (G)
    float* __restrict__ gates,           // (B, G) scratch
    float* __restrict__ c_buf,           // (B, HID) cell state
    _Float16* __restrict__ h16,          // (B, HID) hidden state (f16)
    _Float16* __restrict__ out16,        // layer0: h1_all (T*B,HID) or null
    float* __restrict__ outf,            // layer1: d_out h2 or null
    unsigned* __restrict__ bar) {
  const int tid  = threadIdx.x;
  const int wave = tid >> 5;
  const int lane = tid & 31;
  const int lrow = lane & 15;
  const int aLo  = (lane < 16) ? 0 : 8;
  const int bLo  = (lane < 16) ? 0 : 8;
  unsigned phase = 0;

  // zero-init h and c (workspace is poisoned between runs)
  for (int i = blockIdx.x * 256 + tid; i < BATCH * HIDN; i += NBLK * 256) {
    c_buf[i] = 0.0f;
    h16[i]   = (_Float16)0.0f;
  }
  grid_sync(bar, &phase);

  __shared__ float cin_s[NCHK], cfg_s[NCHK];

  const int NPAIR = (GG / 32) * (BATCH / 16);   // 129 * 4 = 516 strip units

  for (int t = 0; t < T_STEPS; ++t) {
    // ---- Phase 1: gates = t_in[t] + h @ W_hh + b_hh ----
    const float* tin = t_in + (size_t)t * BATCH * GG;
    for (int p = blockIdx.x * 8 + wave; p < NPAIR; p += NBLK * 8) {
      const int m0 = (p / (GG / 32)) * 16;
      const int n0 = (p % (GG / 32)) * 32;
      v8f acc0 = {}, acc1 = {};
      const _Float16* Arow  = h16   + (size_t)(m0 + lrow) * HIDN;
      const _Float16* Brow0 = Wt_hh + (size_t)(n0 + lrow) * HIDN;
      const _Float16* Brow1 = Brow0 + (size_t)16 * HIDN;
#pragma unroll 4
      for (int k0 = 0; k0 < HIDN; k0 += 32) {
        union { v16h v; float4 f[2]; } a, b0, b1;
        a.f[0]  = *(const float4*)(Arow + k0 + aLo);
        a.f[1]  = *(const float4*)(Arow + k0 + aLo + 16);
        b0.f[0] = *(const float4*)(Brow0 + k0 + bLo);
        b0.f[1] = *(const float4*)(Brow0 + k0 + bLo + 16);
        b1.f[0] = *(const float4*)(Brow1 + k0 + bLo);
        b1.f[1] = *(const float4*)(Brow1 + k0 + bLo + 16);
        acc0 = __builtin_amdgcn_wmma_f32_16x16x32_f16(false, a.v, false, b0.v,
                                                      (short)0, acc0, false, false);
        acc1 = __builtin_amdgcn_wmma_f32_16x16x32_f16(false, a.v, false, b1.v,
                                                      (short)0, acc1, false, false);
      }
      const int col    = n0 + lrow;
      const int rowOff = m0 + ((lane < 16) ? 0 : 8);
      const float bv0  = b_hh[col];
      const float bv1  = b_hh[col + 16];
#pragma unroll
      for (int r = 0; r < 8; ++r) {
        size_t row = (size_t)(rowOff + r);
        gates[row * GG + col]      = acc0[r] + tin[row * GG + col] + bv0;
        gates[row * GG + col + 16] = acc1[r] + tin[row * GG + col + 16] + bv1;
      }
    }
    grid_sync(bar, &phase);

    // ---- Phase 2: ON-LSTM pointwise; one block per batch row ----
    const int r = blockIdx.x;  // NBLK == BATCH
    if (tid < 32) {
      const float* g = gates + (size_t)r * GG + ((tid < 16) ? 0 : NCHK);
      const int i = tid & 15;
      float m = g[0];
#pragma unroll
      for (int j = 1; j < 16; ++j) m = fmaxf(m, g[j]);
      float s = 0.0f, p = 0.0f;
#pragma unroll
      for (int j = 0; j < 16; ++j) {
        float e = __expf(g[j] - m);
        s += e;
        if (j <= i) p += e;
      }
      float v = p / s;
      if (tid < 16) cin_s[i] = 1.0f - v;   // 1 - cumsoftmax
      else          cfg_s[i] = v;          // cumsoftmax
    }
    __syncthreads();

    const float* gr = gates + (size_t)r * GG + 2 * NCHK;
#pragma unroll
    for (int u = 0; u < 4; ++u) {
      const int j = tid * 4 + u;          // hidden unit 0..1023
      const int n = j >> 6;               // chunk
      float og = sigm(gr[j]);
      float ce = tanhf(gr[1024 + j]);
      float ig = sigm(gr[2048 + j]);
      float fg = sigm(gr[3072 + j]);
      float ci = cin_s[n], cf = cfg_s[n];
      float ov = cf * ci;
      float f  = fg * ov + (cf - ov);
      float in = ig * ov + (ci - ov);
      float cy = f * c_buf[(size_t)r * HIDN + j] + in * ce;
      float hy = og * tanhf(cy);
      c_buf[(size_t)r * HIDN + j] = cy;
      h16[(size_t)r * HIDN + j]   = (_Float16)hy;
      if (out16) out16[((size_t)t * BATCH + r) * HIDN + j] = (_Float16)hy;
      else       outf[((size_t)t * BATCH + r) * HIDN + j]  = hy;
    }
    __syncthreads();          // protect cin_s/cfg_s before next step
    grid_sync(bar, &phase);
  }
}

// ---------------------------------------------------------------------------
extern "C" void kernel_launch(void* const* d_in, const int* in_sizes, int n_in,
                              void* d_out, int out_size, void* d_ws, size_t ws_size,
                              hipStream_t stream) {
  const int*   tokens = (const int*)  d_in[0];
  const float* emb    = (const float*)d_in[1];
  const float* W_ih0  = (const float*)d_in[2];
  const float* b_ih0  = (const float*)d_in[3];
  const float* W_hh0  = (const float*)d_in[4];
  const float* b_hh0  = (const float*)d_in[5];
  const float* W_ih1  = (const float*)d_in[6];
  const float* b_ih1  = (const float*)d_in[7];
  const float* W_hh1  = (const float*)d_in[8];
  const float* b_hh1  = (const float*)d_in[9];

  float* out_h    = (float*)d_out;                               // (T,B,HID)
  float* out_mask = out_h + (size_t)T_STEPS * BATCH * HIDN;      // (T,B)

  char* ws = (char*)d_ws;
  size_t off = 0;
  auto take = [&](size_t bytes) -> char* {
    char* p = ws + off;
    off = (off + bytes + 255) & ~(size_t)255;
    return p;
  };
  _Float16* wt_ih0 = (_Float16*)take((size_t)GG * HIDN * 2);
  _Float16* wt_hh0 = (_Float16*)take((size_t)GG * HIDN * 2);
  _Float16* wt_ih1 = (_Float16*)take((size_t)GG * HIDN * 2);
  _Float16* wt_hh1 = (_Float16*)take((size_t)GG * HIDN * 2);
  _Float16* x16    = (_Float16*)take((size_t)T_STEPS * BATCH * EMBD * 2);
  _Float16* h1_all = (_Float16*)take((size_t)T_STEPS * BATCH * HIDN * 2);
  float*    t_in   = (float*)   take((size_t)T_STEPS * BATCH * GG * 4);
  float*    gates  = (float*)   take((size_t)BATCH * GG * 4);
  float*    c_buf  = (float*)   take((size_t)BATCH * HIDN * 4);
  _Float16* h16    = (_Float16*)take((size_t)BATCH * HIDN * 2);
  unsigned* bar    = (unsigned*)take(256);

  const dim3 tpb(256);
  const dim3 gT(GG / 32, HIDN / 32);                 // (129, 32)
  transpose_cast<<<gT, tpb, 0, stream>>>(W_ih0, wt_ih0, HIDN, GG);
  transpose_cast<<<gT, tpb, 0, stream>>>(W_hh0, wt_hh0, HIDN, GG);
  transpose_cast<<<gT, tpb, 0, stream>>>(W_ih1, wt_ih1, HIDN, GG);
  transpose_cast<<<gT, tpb, 0, stream>>>(W_hh1, wt_hh1, HIDN, GG);

  embed_cast<<<dim3(T_STEPS * BATCH), tpb, 0, stream>>>(tokens, emb, x16, out_mask);

  const dim3 gG((T_STEPS * BATCH) / 128, GG / 48);   // (128, 86)
  gemm_wmma<<<gG, tpb, 0, stream>>>(x16, wt_ih0, b_ih0, t_in);

  zero_bar<<<1, 32, 0, stream>>>(bar);
  onlstm_recur<<<NBLK, tpb, 0, stream>>>(t_in, wt_hh0, b_hh0, gates, c_buf, h16,
                                         h1_all, nullptr, bar);

  gemm_wmma<<<gG, tpb, 0, stream>>>(h1_all, wt_ih1, b_ih1, t_in);

  zero_bar<<<1, 32, 0, stream>>>(bar);
  onlstm_recur<<<NBLK, tpb, 0, stream>>>(t_in, wt_hh1, b_hh1, gates, c_buf, h16,
                                         nullptr, out_h, bar);
}
